// OptimizedRCN_57561151701204
// MI455X (gfx1250) — matrix operand order repaired and checked by reading, not compiled, not run on hardware
//
#include <hip/hip_runtime.h>
#include <cstdint>

// ---------------------------------------------------------------------------
// Shapes (fixed by the reference): B=8 P=20 T=32 D=128 H=256 R=256
// ---------------------------------------------------------------------------
#define NBQ 8
#define NPQ 20
#define NTQ 32
#define NDQ 128
#define NHQ 256
#define NRQ 256
#define NROW 5120           // B*P*T == B*T*P rows

typedef __attribute__((ext_vector_type(16))) _Float16 v16h;
typedef __attribute__((ext_vector_type(8)))  _Float16 v8h;
typedef __attribute__((ext_vector_type(8)))  float    v8f;
typedef __attribute__((ext_vector_type(4)))  unsigned int u32x4;
typedef __attribute__((ext_vector_type(8)))  int          i32x8;
typedef __attribute__((ext_vector_type(4)))  int          i32x4;

// ---------------------------------------------------------------------------
// Workspace layout.  Halves first, then floats (256B aligned).
// ---------------------------------------------------------------------------
static constexpr size_t H_PW   = 0;                                // 128x256
static constexpr size_t H_SW1  = H_PW  + (size_t)128*256;          // 544x128
static constexpr size_t H_SW2  = H_SW1 + (size_t)544*128;          // 128x256
static constexpr size_t H_TW1  = H_SW2 + (size_t)128*256;          // 512x128
static constexpr size_t H_TW2  = H_TW1 + (size_t)512*128;          // 128x256
static constexpr size_t H_IW1  = H_TW2 + (size_t)128*256;          // 512x128
static constexpr size_t H_IW2  = H_IW1 + (size_t)512*128;          // 128x256
static constexpr size_t H_FW0  = H_IW2 + (size_t)128*256;          // 256x256
static constexpr size_t H_FW1  = H_FW0 + (size_t)256*256;
static constexpr size_t H_FW2  = H_FW1 + (size_t)256*256;
static constexpr size_t H_WP1  = H_FW2 + (size_t)256*256;          // 256x64
static constexpr size_t H_LWS  = H_WP1 + (size_t)256*64;           // 3x 256x256
static constexpr size_t H_LWR  = H_LWS + (size_t)3*256*256;        // 3x 512x256
static constexpr size_t H_OW   = H_LWR + (size_t)3*512*256;        // 256x128
static constexpr size_t H_FEAT = H_OW  + (size_t)256*128;          // feat f16 (B,T,P,H)
static constexpr size_t H_VEL  = H_FEAT + (size_t)NROW*256;        // vel  f16 (B,T,P,H)
static constexpr size_t H_END  = H_VEL  + (size_t)NROW*256;

static constexpr size_t F_BASE = ((H_END*2 + 255) / 256) * 256;    // byte offset
static constexpr size_t F_CONV = 0;                                // 160*128*32
static constexpr size_t F_ST   = F_CONV + (size_t)160*128*32;      // 768 stats
static constexpr size_t F_YPRE = F_ST + 768;                       // 5120*256
static constexpr size_t F_AGG  = F_YPRE + (size_t)NROW*256;        // 5120*256

#define ST_B1S 0
#define ST_B1Q 128
#define ST_B2S 256
#define ST_B2Q 512

// ---------------------------------------------------------------------------
// TDM: 1-D contiguous tile load (global -> LDS) via Tensor Data Mover.
// nelem8 = number of 8-byte elements.  D# packing per CDNA5 ISA 8.3/8.4.
// This toolchain exposes the 6-arg builtin:
//   (u32x4 g0, i32x8 g1, i32x4 g2, i32x4 g3, i32x8 extra, i32 cpol)
// ---------------------------------------------------------------------------
__device__ __forceinline__ void tdm_load_1d(unsigned lds_off, const void* gptr,
                                            unsigned nelem8) {
  unsigned long long ga = (unsigned long long)gptr;
  u32x4 g0;
  g0[0] = 1u;                                        // count=1 (valid), normal mode
  g0[1] = lds_off;                                   // lds_addr (bytes)
  g0[2] = (unsigned)(ga & 0xffffffffu);              // global_addr[31:0]
  g0[3] = (unsigned)((ga >> 32) & 0x01ffffffu)       // global_addr[56:32]
        | 0x80000000u;                               // type=2 ("image") bits127:126
  i32x8 g1;
  g1[0] = (3 << 16);                                 // workgroup_mask=0, data_size=8B
  g1[1] = (int)((nelem8 & 0xffffu) << 16);           // tensor_dim0[15:0]  @bits63:48
  g1[2] = (int)((nelem8 >> 16) & 0xffffu)            // tensor_dim0[31:16] @bits79:64
        | (1 << 16);                                 // tensor_dim1=1      @bits111:80
  g1[3] = (int)((nelem8 & 0xffffu) << 16);           // tile_dim0          @bits127:112
  g1[4] = 1;                                         // tile_dim1=1        @bits143:128
  g1[5] = (int)nelem8;                               // dim0_stride[31:0]  @bits191:160
  g1[6] = 0;
  g1[7] = 0;
  i32x4 gz  = {0, 0, 0, 0};
  i32x8 gz8 = {0, 0, 0, 0, 0, 0, 0, 0};
  __builtin_amdgcn_tensor_load_to_lds(g0, g1, gz, gz, gz8, 0);
}

// ---------------------------------------------------------------------------
// Weight packer: fp32 (K x N) row-major  ->  f16 WMMA B-operand tiles.
// Tile (kt,nt): lane<16 -> col n=nt*16+lane, k=kt*32+0..15;
//               lane>=16 -> col n=nt*16+lane-16, k=kt*32+16..31.
// ---------------------------------------------------------------------------
__global__ void k_pack_w16(const float* __restrict__ W, _Float16* __restrict__ dst,
                           int K, int N, int Kpad) {
  int ktiles = Kpad >> 5, ntiles = N >> 4;
  int total = ktiles * ntiles * 512;
  for (int x = blockIdx.x * blockDim.x + threadIdx.x; x < total;
       x += gridDim.x * blockDim.x) {
    int idx  = x & 15;
    int lane = (x >> 4) & 31;
    int tile = x >> 9;
    int nt = tile % ntiles, kt = tile / ntiles;
    int n = (nt << 4) + (lane & 15);
    int k = (kt << 5) + ((lane >> 4) << 4) + idx;
    float v = (k < K) ? W[(size_t)k * N + n] : 0.f;
    dst[x] = (_Float16)v;
  }
}

// ---------------------------------------------------------------------------
// Workgroup GEMM: 32 rows (f16 in LDS, stride lda) x packed weights.
// 4 waves; tiles = 2 row-tiles x (N/16) n-tiles; f32 accumulate in WMMA.
// ---------------------------------------------------------------------------
__device__ __forceinline__ void wg_gemm(const _Float16* __restrict__ A, int lda,
                                        int ktiles, const _Float16* __restrict__ Bpk,
                                        int N, float* Cf, _Float16* Ch, int ldc,
                                        const float* __restrict__ bias, float scale,
                                        bool relu, bool accum) {
  const int lane = threadIdx.x & 31;
  const int wave = threadIdx.x >> 5;
  const int ntiles = N >> 4;
  for (int tile = wave; tile < 2 * ntiles; tile += 4) {
    const int rt = tile & 1;
    const int nt = tile >> 1;
    const int m0 = rt << 4;
    v8f acc = {};
    const _Float16* arow = A + (size_t)(m0 + (lane & 15)) * lda + ((lane >> 4) << 4);
    const _Float16* bptr = Bpk + ((size_t)nt * 32 + lane) * 16;
    for (int kt = 0; kt < ktiles; ++kt) {
      v8h lo = *(const v8h*)(arow + kt * 32);
      v8h hi = *(const v8h*)(arow + kt * 32 + 8);
      v16h a;
#pragma unroll
      for (int u = 0; u < 8; ++u) { a[u] = lo[u]; a[u + 8] = hi[u]; }
      v16h b = *(const v16h*)(bptr + (size_t)kt * ntiles * 512);
      if (kt + 1 < ktiles)
        __builtin_prefetch(bptr + (size_t)(kt + 1) * ntiles * 512, 0, 3);
      acc = __builtin_amdgcn_wmma_f32_16x16x32_f16(false, a, false, b, (short)0, acc,
                                                   false, false);
    }
    const int mb = m0 + ((lane >> 4) << 3);
    const int n  = (nt << 4) + (lane & 15);
    const float bv = bias ? bias[n] : 0.f;
#pragma unroll
    for (int u = 0; u < 8; ++u) {
      float x = acc[u] + bv;
      if (relu) x = fmaxf(x, 0.f);
      x *= scale;
      if (Cf) {
        if (accum) Cf[(size_t)(mb + u) * ldc + n] += x;
        else       Cf[(size_t)(mb + u) * ldc + n]  = x;
      }
      if (Ch) Ch[(size_t)(mb + u) * ldc + n] = (_Float16)x;
    }
  }
}

// ---------------------------------------------------------------------------
// K1: depthwise conv (kernel 3, SAME over T) + BN1 channel stats.
// ---------------------------------------------------------------------------
__global__ void k_conv_bn1(const float* __restrict__ feats, const float* __restrict__ dww,
                           const float* __restrict__ dwb, float* __restrict__ conv,
                           float* __restrict__ st) {
  int idx = blockIdx.x * blockDim.x + threadIdx.x;  // 160*128
  int c = idx & 127, n = idx >> 7;
  float w0 = dww[c * 3 + 0], w1 = dww[c * 3 + 1], w2 = dww[c * 3 + 2], bb = dwb[c];
  const float* xp = feats + (size_t)n * NTQ * NDQ + c;
  float s = 0.f, q = 0.f;
  for (int t = 0; t < NTQ; ++t) {
    float xm = (t > 0)  ? xp[(t - 1) * NDQ] : 0.f;
    float x0 = xp[t * NDQ];
    float xp1 = (t < NTQ - 1) ? xp[(t + 1) * NDQ] : 0.f;
    float y = xm * w0 + x0 * w1 + xp1 * w2 + bb;
    conv[((size_t)n * NDQ + c) * NTQ + t] = y;
    s += y; q += y * y;
  }
  atomicAdd(&st[ST_B1S + c], s);
  atomicAdd(&st[ST_B1Q + c], q);
}

// ---------------------------------------------------------------------------
// K2: BN1 apply + ReLU + pointwise WMMA GEMM (K=128 -> H=256) + BN2 stats.
// ---------------------------------------------------------------------------
__global__ void k_pointwise(const float* __restrict__ conv, const float* __restrict__ st,
                            const float* __restrict__ g1, const float* __restrict__ b1,
                            const _Float16* __restrict__ pwpk, const float* __restrict__ pwb,
                            float* __restrict__ ypre, float* __restrict__ stout) {
  __shared__ alignas(16) _Float16 Alds[32 * 128];
  __shared__ alignas(16) float    Clds[32 * 256];
  __shared__ float sc[128], sh[128];
  int tid = threadIdx.x;
  if (tid < 128) {
    float m = st[ST_B1S + tid] * (1.f / 5120.f);
    float v = st[ST_B1Q + tid] * (1.f / 5120.f) - m * m;
    float s = g1[tid] * rsqrtf(v + 1e-5f);
    sc[tid] = s; sh[tid] = b1[tid] - m * s;
  }
  __syncthreads();
  int R0 = blockIdx.x * 32;
  for (int e = tid; e < 32 * 128; e += 128) {
    int c = e & 127, r = e >> 7;
    int R = R0 + r, n = R >> 5, t = R & 31;
    float x = conv[((size_t)n * NDQ + c) * NTQ + t] * sc[c] + sh[c];
    Alds[r * 128 + c] = (_Float16)fmaxf(x, 0.f);
  }
  __syncthreads();
  wg_gemm(Alds, 128, 4, pwpk, 256, Clds, nullptr, 256, nullptr, 1.f, false, false);
  __syncthreads();
  for (int h = tid; h < 256; h += 128) {
    float bb = pwb[h], s = 0.f, q = 0.f;
    for (int r = 0; r < 32; ++r) {
      float v = Clds[r * 256 + h] + bb;
      ypre[(size_t)(R0 + r) * 256 + h] = v;
      s += v; q += v * v;
    }
    atomicAdd(&stout[ST_B2S + h], s);
    atomicAdd(&stout[ST_B2Q + h], q);
  }
}

// ---------------------------------------------------------------------------
// K3: BN2 apply + ReLU -> feat f16 (B,T,P,H); velocities -> f16 (B,T,P,H).
// ---------------------------------------------------------------------------
__global__ void k_bn2_feat(const float* __restrict__ ypre, const float* __restrict__ st,
                           const float* __restrict__ g2, const float* __restrict__ b2,
                           const float* __restrict__ vel, _Float16* __restrict__ feath,
                           _Float16* __restrict__ velh) {
  for (size_t idx = (size_t)blockIdx.x * blockDim.x + threadIdx.x;
       idx < (size_t)NROW * 256; idx += (size_t)gridDim.x * blockDim.x) {
    int h = idx & 255;
    size_t row = idx >> 8;
    int t = row & 31;
    int n = (int)(row >> 5);
    int b = n / NPQ, p = n % NPQ;
    float m = st[ST_B2S + h] * (1.f / 5120.f);
    float v = st[ST_B2Q + h] * (1.f / 5120.f) - m * m;
    float s = g2[h] * rsqrtf(v + 1e-5f);
    float val = fmaxf(ypre[idx] * s + (b2[h] - m * s), 0.f);
    size_t dst = ((((size_t)b * NTQ + t) * NPQ + p) << 8) + h;
    feath[dst] = (_Float16)val;
    velh[dst]  = (_Float16)vel[idx];
  }
}

// ---------------------------------------------------------------------------
// K4: fused pair kernel.  One WG per (b,t,i).  feat/vel rows for the (b,t)
// block are staged into LDS once via the Tensor Data Mover, then all three
// MLPs + fusion + LN + softmax + aggregation run out of LDS with WMMA.
// ---------------------------------------------------------------------------
struct PairArgs {
  const _Float16 *feath, *velh;
  const float    *positions;
  const _Float16 *SW1, *SW2, *TW1, *TW2, *IW1, *IW2, *FW0, *FW1, *FW2, *WP1;
  const float *sb1, *sb2, *tb1, *tb2, *ib1, *ib2, *fb, *lng, *lnb;
  const float *wpb1, *wpW2, *wpb2, *relw;
  float *agg;
};

__global__ __launch_bounds__(128) void k_pair(PairArgs pa) {
  __shared__ alignas(16) _Float16 fstage[NPQ * 256];   // feat rows (b,t,*)
  __shared__ alignas(16) _Float16 vstage[NPQ * 256];   // vel rows  (b,t,*)
  __shared__ alignas(16) _Float16 inbuf[32 * 544];
  __shared__ alignas(16) _Float16 hid[32 * 128];
  __shared__ alignas(16) _Float16 out2[32 * 256];      // reused as LN'd rel (f16)
  __shared__ alignas(16) float    rel[32 * 256];
  __shared__ float sv[32], av[32], red[2];

  const int tid = threadIdx.x;
  const int wg = blockIdx.x;
  const int i = wg % NPQ;
  const int t = (wg / NPQ) & 31;
  const int b = wg / (NPQ * NTQ);

  const _Float16* fbase = pa.feath + ((size_t)(b * NTQ + t) * NPQ) * 256;
  const _Float16* vbase = pa.velh  + ((size_t)(b * NTQ + t) * NPQ) * 256;

  // --- TDM stage: one descriptor per tensor, issued by wave 0 --------------
  if ((tid >> 5) == 0) {
    tdm_load_1d((unsigned)(unsigned long long)(void*)fstage, fbase,
                (NPQ * 256 * 2) / 8);
    tdm_load_1d((unsigned)(unsigned long long)(void*)vstage, vbase,
                (NPQ * 256 * 2) / 8);
    __builtin_amdgcn_s_wait_tensorcnt(0);
  }
  __syncthreads();

  // softmax(rel_w) — 3 scalars, computed redundantly per thread.
  float r0 = pa.relw[0], r1 = pa.relw[1], r2 = pa.relw[2];
  float rmx = fmaxf(r0, fmaxf(r1, r2));
  float e0 = expf(r0 - rmx), e1 = expf(r1 - rmx), e2 = expf(r2 - rmx);
  float rden = 1.f / (e0 + e1 + e2);
  float w0 = e0 * rden, w1 = e1 * rden, w2 = e2 * rden;

  const float pix = pa.positions[((size_t)(b * NPQ + i) * NTQ + t) * 3 + 0];
  const float piy = pa.positions[((size_t)(b * NPQ + i) * NTQ + t) * 3 + 1];
  const float piz = pa.positions[((size_t)(b * NPQ + i) * NTQ + t) * 3 + 2];

  // ---- spatial: [Fi | Fj | geom]  (K=515, pad 544) ------------------------
  for (int e = tid; e < 32 * 544; e += 128) {
    int k = e % 544, j = e / 544;
    float v = 0.f;
    if (j < NPQ) {
      if (k < 256) v = (float)fstage[i * 256 + k];
      else if (k < 512) v = (float)fstage[j * 256 + (k - 256)];
      else if (k < 515) {
        const float* pj = pa.positions + ((size_t)(b * NPQ + j) * NTQ + t) * 3;
        float dx = pj[0] - pix, dy = pj[1] - piy, dz = pj[2] - piz;
        float d2 = dx * dx + dy * dy + dz * dz;
        if (k == 512) v = sqrtf(d2 + 1e-12f);
        else if (k == 513) { bool z = d2 < 1e-20f; v = atan2f(z ? 0.f : dx, z ? 1.f : dz); }
        else v = dy;
      }
    }
    inbuf[e] = (_Float16)v;
  }
  __syncthreads();
  wg_gemm(inbuf, 544, 17, pa.SW1, 128, nullptr, hid, 128, pa.sb1, 1.f, true, false);
  __syncthreads();
  wg_gemm(hid, 128, 4, pa.SW2, 256, nullptr, out2, 256, pa.sb2, w0, false, false);
  __syncthreads();
  wg_gemm(out2, 256, 8, pa.FW0, 256, rel, nullptr, 256, pa.fb, 1.f, false, false);
  __syncthreads();

  // ---- temporal: [Vi | Vj]  (K=512) ---------------------------------------
  for (int e = tid; e < 32 * 512; e += 128) {
    int k = e & 511, j = e >> 9;
    float v = 0.f;
    if (j < NPQ) v = (k < 256) ? (float)vstage[i * 256 + k]
                               : (float)vstage[j * 256 + (k - 256)];
    inbuf[j * 544 + k] = (_Float16)v;
  }
  __syncthreads();
  wg_gemm(inbuf, 544, 16, pa.TW1, 128, nullptr, hid, 128, pa.tb1, 1.f, true, false);
  __syncthreads();
  wg_gemm(hid, 128, 4, pa.TW2, 256, nullptr, out2, 256, pa.tb2, w1, false, false);
  __syncthreads();
  wg_gemm(out2, 256, 8, pa.FW1, 256, rel, nullptr, 256, nullptr, 1.f, false, true);
  __syncthreads();

  // ---- interaction: [fi*fj | Fi+Fj]  (K=512) ------------------------------
  for (int e = tid; e < 32 * 512; e += 128) {
    int k = e & 511, j = e >> 9;
    float v = 0.f;
    if (j < NPQ) {
      float a_ = (float)fstage[i * 256 + (k & 255)];
      float c_ = (float)fstage[j * 256 + (k & 255)];
      v = (k < 256) ? a_ * c_ : a_ + c_;
    }
    inbuf[j * 544 + k] = (_Float16)v;
  }
  __syncthreads();
  wg_gemm(inbuf, 544, 16, pa.IW1, 128, nullptr, hid, 128, pa.ib1, 1.f, true, false);
  __syncthreads();
  wg_gemm(hid, 128, 4, pa.IW2, 256, nullptr, out2, 256, pa.ib2, w2, false, false);
  __syncthreads();
  wg_gemm(out2, 256, 8, pa.FW2, 256, rel, nullptr, 256, nullptr, 1.f, false, true);
  __syncthreads();

  // ---- LayerNorm per row; keep f32 in rel, f16 copy in out2 ---------------
  if (tid < 32) {
    int j = tid;
    float m = 0.f;
    for (int c = 0; c < 256; ++c) m += rel[j * 256 + c];
    m *= (1.f / 256.f);
    float v = 0.f;
    for (int c = 0; c < 256; ++c) { float d = rel[j * 256 + c] - m; v += d * d; }
    float inv = rsqrtf(v * (1.f / 256.f) + 1e-5f);
    for (int c = 0; c < 256; ++c) {
      float x = (rel[j * 256 + c] - m) * inv * pa.lng[c] + pa.lnb[c];
      rel[j * 256 + c] = x;
      out2[j * 256 + c] = (_Float16)x;
    }
  }
  __syncthreads();

  // ---- wp MLP: 256 -> 64 (WMMA, relu) -> 1 (dot) --------------------------
  wg_gemm(out2, 256, 8, pa.WP1, 64, nullptr, hid, 128, pa.wpb1, 1.f, true, false);
  __syncthreads();
  if (tid < 32) {
    float s = pa.wpb2[0];
    for (int k = 0; k < 64; ++k) s += (float)hid[tid * 128 + k] * pa.wpW2[k];
    sv[tid] = s;
  }
  __syncthreads();
  if (tid == 0) {
    float mx = -1e30f;
    for (int j = 0; j < NPQ; ++j) mx = fmaxf(mx, sv[j]);
    float den = 0.f;
    for (int j = 0; j < NPQ; ++j) den += expf(sv[j] - mx);
    red[0] = mx; red[1] = 1.f / den;
  }
  __syncthreads();
  if (tid < NPQ) av[tid] = expf(sv[tid] - red[0]) * red[1];
  __syncthreads();

  // ---- agg[i,c] = mean_j rel[j,c] * a[j] ----------------------------------
  for (int c = tid; c < 256; c += 128) {
    float s = 0.f;
    for (int j = 0; j < NPQ; ++j) s += rel[j * 256 + c] * av[j];
    pa.agg[((size_t)(b * NTQ + t) * NPQ + i) * 256 + c] = s * (1.f / NPQ);
  }
}

// ---------------------------------------------------------------------------
// K5: node layers (3x: node@Ws + [node|agg]@Wr, LN, relu) + output GEMM.
// ---------------------------------------------------------------------------
struct NodeArgs {
  const _Float16 *feath;
  const float    *agg;
  const _Float16 *LWS, *LWR, *OW;
  const float    *bs[3], *br[3], *g[3], *bb[3];
  const float    *outb;
  float          *out;
};

__global__ __launch_bounds__(128) void k_node(NodeArgs na) {
  __shared__ alignas(16) _Float16 cat[32 * 512];  // [node(256) | agg(256)]
  __shared__ alignas(16) float    acc[32 * 256];
  int tid = threadIdx.x;
  int R0 = blockIdx.x * 32;

  for (int e = tid; e < 32 * 512; e += 128) {
    int k = e & 511, r = e >> 9;
    size_t R = (size_t)R0 + r;
    cat[e] = (k < 256) ? na.feath[R * 256 + k]
                       : (_Float16)na.agg[R * 256 + (k - 256)];
  }
  __syncthreads();

  for (int l = 0; l < 3; ++l) {
    wg_gemm(cat, 512, 8,  na.LWS + (size_t)l * 256 * 256, 256, acc, nullptr, 256,
            na.bs[l], 1.f, false, false);
    wg_gemm(cat, 512, 16, na.LWR + (size_t)l * 512 * 256, 256, acc, nullptr, 256,
            na.br[l], 1.f, false, true);
    __syncthreads();
    if (tid < 32) {
      int r = tid;
      float m = 0.f;
      for (int c = 0; c < 256; ++c) m += acc[r * 256 + c];
      m *= (1.f / 256.f);
      float v = 0.f;
      for (int c = 0; c < 256; ++c) { float d = acc[r * 256 + c] - m; v += d * d; }
      float inv = rsqrtf(v * (1.f / 256.f) + 1e-5f);
      for (int c = 0; c < 256; ++c) {
        float x = (acc[r * 256 + c] - m) * inv * na.g[l][c] + na.bb[l][c];
        cat[r * 512 + c] = (_Float16)fmaxf(x, 0.f);
      }
    }
    __syncthreads();
  }

  wg_gemm(cat, 512, 8, na.OW, 128, acc, nullptr, 256, na.outb, 1.f, false, false);
  __syncthreads();
  for (int e = tid; e < 32 * 128; e += 128) {
    int d = e & 127, r = e >> 7;
    int R = R0 + r;
    int p = R % NPQ, tq = (R / NPQ) % NTQ, bq = R / (NPQ * NTQ);
    na.out[(((size_t)bq * NPQ + p) * NTQ + tq) * NDQ + d] = acc[r * 256 + d];
  }
}

// ---------------------------------------------------------------------------
// Host launcher.  d_in order: features, positions, velocities, then params
// flattened as a JAX pytree (dict keys sorted; 'W' < lowercase).
// ---------------------------------------------------------------------------
extern "C" void kernel_launch(void* const* d_in, const int* in_sizes, int n_in,
                              void* d_out, int out_size, void* d_ws, size_t ws_size,
                              hipStream_t stream) {
  (void)in_sizes; (void)n_in; (void)out_size; (void)ws_size;
  const float* features   = (const float*)d_in[0];
  const float* positions  = (const float*)d_in[1];
  const float* velocities = (const float*)d_in[2];
  const float* bn1_b = (const float*)d_in[3];
  const float* bn1_g = (const float*)d_in[4];
  const float* bn2_b = (const float*)d_in[5];
  const float* bn2_g = (const float*)d_in[6];
  const float* dw_b  = (const float*)d_in[7];
  const float* dw_w  = (const float*)d_in[8];
  const float* fusion_W = (const float*)d_in[9];
  const float* fusion_b = (const float*)d_in[10];
  const float* i_W1 = (const float*)d_in[11];
  const float* i_W2 = (const float*)d_in[12];
  const float* i_b1 = (const float*)d_in[13];
  const float* i_b2 = (const float*)d_in[14];
  const float* l_Wr[3]; const float* l_Ws[3]; const float* l_b[3];
  const float* l_br[3]; const float* l_bs[3]; const float* l_g[3];
  for (int l = 0; l < 3; ++l) {
    l_Wr[l] = (const float*)d_in[15 + 6 * l + 0];
    l_Ws[l] = (const float*)d_in[15 + 6 * l + 1];
    l_b[l]  = (const float*)d_in[15 + 6 * l + 2];
    l_br[l] = (const float*)d_in[15 + 6 * l + 3];
    l_bs[l] = (const float*)d_in[15 + 6 * l + 4];
    l_g[l]  = (const float*)d_in[15 + 6 * l + 5];
  }
  const float* ln_b  = (const float*)d_in[33];
  const float* ln_g  = (const float*)d_in[34];
  const float* out_W = (const float*)d_in[35];
  const float* out_b = (const float*)d_in[36];
  const float* pw_b  = (const float*)d_in[37];
  const float* pw_w  = (const float*)d_in[38];
  const float* rel_w = (const float*)d_in[39];
  const float* s_W1  = (const float*)d_in[40];
  const float* s_W2  = (const float*)d_in[41];
  const float* s_b1  = (const float*)d_in[42];
  const float* s_b2  = (const float*)d_in[43];
  const float* t_W1  = (const float*)d_in[44];
  const float* t_W2  = (const float*)d_in[45];
  const float* t_b1  = (const float*)d_in[46];
  const float* t_b2  = (const float*)d_in[47];
  const float* wp_W1 = (const float*)d_in[48];
  const float* wp_W2 = (const float*)d_in[49];
  const float* wp_b1 = (const float*)d_in[50];
  const float* wp_b2 = (const float*)d_in[51];

  _Float16* hb = (_Float16*)d_ws;
  float* fb    = (float*)((char*)d_ws + F_BASE);
  float* st    = fb + F_ST;

  auto pack = [&](const float* src, size_t hoff, int K, int N, int Kpad) {
    int total = (Kpad / 32) * (N / 16) * 512;
    k_pack_w16<<<(total + 255) / 256, 256, 0, stream>>>(src, hb + hoff, K, N, Kpad);
  };
  pack(pw_w, H_PW, 128, 256, 128);
  pack(s_W1, H_SW1, 515, 128, 544);
  pack(s_W2, H_SW2, 128, 256, 128);
  pack(t_W1, H_TW1, 512, 128, 512);
  pack(t_W2, H_TW2, 128, 256, 128);
  pack(i_W1, H_IW1, 512, 128, 512);
  pack(i_W2, H_IW2, 128, 256, 128);
  pack(fusion_W,             H_FW0, 256, 256, 256);
  pack(fusion_W + 256 * 256, H_FW1, 256, 256, 256);
  pack(fusion_W + 512 * 256, H_FW2, 256, 256, 256);
  pack(wp_W1, H_WP1, 256, 64, 256);
  for (int l = 0; l < 3; ++l) {
    pack(l_Ws[l], H_LWS + (size_t)l * 256 * 256, 256, 256, 256);
    pack(l_Wr[l], H_LWR + (size_t)l * 512 * 256, 512, 256, 512);
  }
  pack(out_W, H_OW, 256, 128, 256);

  (void)hipMemsetAsync(st, 0, 768 * sizeof(float), stream);

  k_conv_bn1<<<160, 128, 0, stream>>>(features, dw_w, dw_b, fb + F_CONV, st);
  k_pointwise<<<160, 128, 0, stream>>>(fb + F_CONV, st, bn1_g, bn1_b,
                                       hb + H_PW, pw_b, fb + F_YPRE, st);
  k_bn2_feat<<<640, 256, 0, stream>>>(fb + F_YPRE, st, bn2_g, bn2_b, velocities,
                                      hb + H_FEAT, hb + H_VEL);

  PairArgs pa;
  pa.feath = hb + H_FEAT; pa.velh = hb + H_VEL; pa.positions = positions;
  pa.SW1 = hb + H_SW1; pa.SW2 = hb + H_SW2;
  pa.TW1 = hb + H_TW1; pa.TW2 = hb + H_TW2;
  pa.IW1 = hb + H_IW1; pa.IW2 = hb + H_IW2;
  pa.FW0 = hb + H_FW0; pa.FW1 = hb + H_FW1; pa.FW2 = hb + H_FW2;
  pa.WP1 = hb + H_WP1;
  pa.sb1 = s_b1; pa.sb2 = s_b2; pa.tb1 = t_b1; pa.tb2 = t_b2;
  pa.ib1 = i_b1; pa.ib2 = i_b2; pa.fb = fusion_b; pa.lng = ln_g; pa.lnb = ln_b;
  pa.wpb1 = wp_b1; pa.wpW2 = wp_W2; pa.wpb2 = wp_b2; pa.relw = rel_w;
  pa.agg = fb + F_AGG;
  k_pair<<<NBQ * NTQ * NPQ, 128, 0, stream>>>(pa);

  NodeArgs na;
  na.feath = hb + H_FEAT; na.agg = fb + F_AGG;
  na.LWS = hb + H_LWS; na.LWR = hb + H_LWR; na.OW = hb + H_OW;
  for (int l = 0; l < 3; ++l) {
    na.bs[l] = l_bs[l]; na.br[l] = l_br[l]; na.g[l] = l_g[l]; na.bb[l] = l_b[l];
  }
  na.outb = out_b; na.out = (float*)d_out;
  k_node<<<NROW / 32, 128, 0, stream>>>(na);
}
